// LocalBlock_24172075942147
// MI455X (gfx1250) — matrix operand order repaired and checked by reading, not compiled, not run on hardware
//
#include <hip/hip_runtime.h>

// ---------------------------------------------------------------------------
// CDNA5 (gfx1250) wave32 WMMA GNN pipeline.
// All dense contractions use v_wmma_f32_16x16x32_bf16 (f32 accum).
// All gather/scatter indexing uses u32 element offsets so the compiler can
// emit SGPR-base + 32-bit-VGPR-offset (GVS) global addressing.
// ---------------------------------------------------------------------------

typedef __attribute__((ext_vector_type(16))) __bf16 v16bf;
typedef __attribute__((ext_vector_type(8)))  float  v8f;

union BFrag  { v16bf v; unsigned u[8]; };
union PackBF { __bf16 h[2]; unsigned u; };

__device__ __forceinline__ unsigned pack_bf16(float lo, float hi) {
  PackBF p; p.h[0] = (__bf16)lo; p.h[1] = (__bf16)hi;   // native v_cvt_pk_bf16_f32
  return p.u;
}

// A matrix fragment: 16x32 bf16, source row-major f32 with leading dim ld.
// Layout: lanes 0-15 hold M=lane, K = 0..15;
//         lanes 16-31 hold M=lane-16, K = 16..31.
__device__ __forceinline__ v16bf load_a_bf16(const float* S, int ld, int k0, int lane) {
  int m = lane & 15, half = lane >> 4;
  const float* row = S + (unsigned)m * (unsigned)ld;
  BFrag r;
#pragma unroll
  for (int p = 0; p < 8; ++p) {
    int e0 = 2 * p;
    int k  = k0 + e0 + 16 * half;
    r.u[p] = pack_bf16(row[k], row[k + 1]);
  }
  return r.v;
}

// B matrix fragment: 32x16 bf16 from row-major W (K x NC), column block n0.
// Lane = column (lane&15); lanes 0-15 carry K=k0..k0+15, lanes 16-31 K=k0+16..k0+31.
__device__ __forceinline__ v16bf load_b_bf16(const float* W, int ldw, int k0, int n0, int lane) {
  int n  = n0 + (lane & 15);
  int kb = k0 + (lane >> 4) * 16;
  BFrag r;
#pragma unroll
  for (int p = 0; p < 8; ++p)
    r.u[p] = pack_bf16(W[(unsigned)(kb + 2 * p) * (unsigned)ldw + n],
                       W[(unsigned)(kb + 2 * p + 1) * (unsigned)ldw + n]);
  return r.v;
}

// Pre-convert all B fragments of a K=32 weight matrix (32 x NC) into LDS,
// laid out [NC/16][lane][8] u32 so a fragment load is 2x ds_load_b128.
__device__ __forceinline__ void stage_b32_frags(const float* W, int NC, unsigned* sWB,
                                                int tid, int nthreads) {
  int total = (NC / 16) * 32 * 8;
  for (int i = tid; i < total; i += nthreads) {
    int p = i & 7, lane = (i >> 3) & 31, nb = i >> 8;
    int n  = nb * 16 + (lane & 15);
    int kb = (lane >> 4) * 16 + 2 * p;
    sWB[i] = pack_bf16(W[(unsigned)kb * (unsigned)NC + n],
                       W[(unsigned)(kb + 1) * (unsigned)NC + n]);
  }
}
__device__ __forceinline__ v16bf load_b_lds(const unsigned* sWB, int nb, int lane) {
  const unsigned* s = sWB + ((unsigned)nb * 32 + (unsigned)lane) * 8;
  BFrag r;
#pragma unroll
  for (int i = 0; i < 8; ++i) r.u[i] = s[i];
  return r.v;
}

__device__ __forceinline__ v8f wmma32(v16bf a, v16bf b, v8f c) {
  return __builtin_amdgcn_wmma_f32_16x16x32_bf16(false, a, false, b, (short)0, c, false, false);
}

// order-preserving float <-> uint key (for atomicMax-based segment max)
__device__ __forceinline__ unsigned fkey(float f) {
  unsigned b = __float_as_uint(f);
  return (b & 0x80000000u) ? ~b : (b | 0x80000000u);
}
__device__ __forceinline__ float funkey(unsigned k) {
  unsigned b = (k & 0x80000000u) ? (k & 0x7FFFFFFFu) : ~k;
  return __uint_as_float(b);
}

// ---------------------------------------------------------------------------
// elementwise helpers
// ---------------------------------------------------------------------------
__global__ void zero_kernel(float* p, int n) {
  int i = blockIdx.x * blockDim.x + threadIdx.x;
  if (i < n) p[i] = 0.0f;
}
__global__ void gine_h_kernel(const float* __restrict__ x, const float* __restrict__ agg,
                              const float* __restrict__ eps, float* __restrict__ h, int n) {
  int i = blockIdx.x * blockDim.x + threadIdx.x;
  if (i < n) h[i] = (1.0f + eps[0]) * x[i] + agg[i];
}
__global__ void add_kernel(const float* __restrict__ a, const float* __restrict__ b,
                           float* __restrict__ o, int n) {
  int i = blockIdx.x * blockDim.x + threadIdx.x;
  if (i < n) o[i] = a[i] + b[i];
}
__global__ void bias_out_kernel(const float* __restrict__ a, const float* __restrict__ bias,
                                float* __restrict__ o, int n) {
  int i = blockIdx.x * blockDim.x + threadIdx.x;
  if (i < n) o[i] = a[i] + bias[i & 63];
}

// ---------------------------------------------------------------------------
// GINE edge aggregate: agg[dst] += relu(x[src] + (ea @ we + be))
// wave = 16 edges; edge-lin is K=32 -> one WMMA per 16-col tile.
// ---------------------------------------------------------------------------
__global__ __launch_bounds__(256) void gine_edge_kernel(
    const float* __restrict__ x, const float* __restrict__ ea,
    const float* __restrict__ we /*32x64*/, const float* __restrict__ be,
    const int* __restrict__ src, const int* __restrict__ dst,
    float* __restrict__ agg, int E_) {
  __shared__ unsigned sWB[4 * 32 * 8];
  __shared__ float sB[64];
  __shared__ float sEA[8][16 * 32];
  __shared__ int   sSrc[8][16], sDst[8][16];
  int tid = threadIdx.x;
  stage_b32_frags(we, 64, sWB, tid, 256);
  if (tid < 64) sB[tid] = be[tid];
  __syncthreads();

  int wave = tid >> 5, lane = tid & 31;
  int ebase = (blockIdx.x * 8 + wave) * 16;
  if (ebase >= E_) return;

  for (int i = lane; i < 16 * 32; i += 32) sEA[wave][i] = ea[(size_t)ebase * 32 + i];
  if (lane < 16) { sSrc[wave][lane] = src[ebase + lane]; sDst[wave][lane] = dst[ebase + lane]; }

  v16bf a = load_a_bf16(sEA[wave], 32, 0, lane);
  int half = lane >> 4, nin = lane & 15;
  unsigned os[8], od[8];
#pragma unroll
  for (int r = 0; r < 8; ++r) {
    os[r] = (unsigned)sSrc[wave][r + 8 * half] * 64u + (unsigned)nin;
    od[r] = (unsigned)sDst[wave][r + 8 * half] * 64u + (unsigned)nin;
  }
#pragma unroll
  for (int nb = 0; nb < 4; ++nb) {
    unsigned n0 = nb * 16;
    v16bf b = load_b_lds(sWB, nb, lane);
    float bias = sB[n0 + nin];
    v8f c;
#pragma unroll
    for (int r = 0; r < 8; ++r) c[r] = bias;
    c = wmma32(a, b, c);
    float xv[8];
#pragma unroll
    for (int r = 0; r < 8; ++r) xv[r] = x[os[r] + n0];
#pragma unroll
    for (int r = 0; r < 8; ++r)
      atomicAdd(&agg[od[r] + n0], fmaxf(c[r] + xv[r], 0.0f));
  }
}

// ---------------------------------------------------------------------------
// Generic node linear: out = [relu](X @ W + bias), X is Nrows x K row-major,
// W is K x NC row-major. Wave computes 16 rows x NC cols.
// ---------------------------------------------------------------------------
template <int K, int NW>
__global__ __launch_bounds__(NW * 32) void node_linear_kernel(
    const float* __restrict__ X, const float* __restrict__ W, const float* __restrict__ bias,
    float* __restrict__ out, int NC, int doRelu, int Nrows) {
  __shared__ float sX[NW][16 * K];
  int wave = threadIdx.x >> 5, lane = threadIdx.x & 31;
  int rbase = (blockIdx.x * NW + wave) * 16;
  if (rbase >= Nrows) return;
  for (int i = lane; i < 16 * K; i += 32) sX[wave][i] = X[(size_t)rbase * K + i];
  v16bf a[K / 32];
#pragma unroll
  for (int kk = 0; kk < K / 32; ++kk) a[kk] = load_a_bf16(sX[wave], K, kk * 32, lane);
  int half = lane >> 4, nin = lane & 15;
  for (int n0 = 0; n0 < NC; n0 += 16) {
    float bv = bias ? bias[n0 + nin] : 0.0f;
    v8f c;
#pragma unroll
    for (int r = 0; r < 8; ++r) c[r] = bv;
#pragma unroll
    for (int kk = 0; kk < K / 32; ++kk) {
      v16bf b = load_b_bf16(W, NC, kk * 32, n0, lane);
      c = wmma32(a[kk], b, c);
    }
#pragma unroll
    for (int r = 0; r < 8; ++r) {
      float v = c[r];
      if (doRelu) v = fmaxf(v, 0.0f);
      out[(unsigned)(rbase + r + 8 * half) * (unsigned)NC + (unsigned)(n0 + nin)] = v;
    }
  }
}

// ---------------------------------------------------------------------------
// TransformerConv pass 1: logits alpha[e,h] = <q[dst,h], k[src,h]+e3[e,h]>/8
// e3 = ea @ we3 recomputed tile-by-tile with WMMA (K=32, 16 col tiles).
// ---------------------------------------------------------------------------
__global__ __launch_bounds__(256) void trans_logit_kernel(
    const float* __restrict__ q, const float* __restrict__ k,
    const float* __restrict__ ea, const float* __restrict__ we3 /*32x256*/,
    const int* __restrict__ src, const int* __restrict__ dst,
    float* __restrict__ logit /*E*4*/, unsigned* __restrict__ nmax /*N*4*/, int E_) {
  __shared__ unsigned sWB[16 * 32 * 8];
  __shared__ float sEA[8][16 * 32];
  __shared__ int   sSrc[8][16], sDst[8][16];
  __shared__ float sAcc[8][64];
  int tid = threadIdx.x;
  stage_b32_frags(we3, 256, sWB, tid, 256);
  __syncthreads();

  int wave = tid >> 5, lane = tid & 31;
  int ebase = (blockIdx.x * 8 + wave) * 16;
  if (ebase >= E_) return;
  for (int i = lane; i < 16 * 32; i += 32) sEA[wave][i] = ea[(size_t)ebase * 32 + i];
  if (lane < 16) { sSrc[wave][lane] = src[ebase + lane]; sDst[wave][lane] = dst[ebase + lane]; }
  for (int i = lane; i < 64; i += 32) sAcc[wave][i] = 0.0f;

  v16bf a = load_a_bf16(sEA[wave], 32, 0, lane);
  int half = lane >> 4, nin = lane & 15;
  unsigned os[8], od[8];
#pragma unroll
  for (int r = 0; r < 8; ++r) {
    os[r] = (unsigned)sSrc[wave][r + 8 * half] * 256u + (unsigned)nin;
    od[r] = (unsigned)sDst[wave][r + 8 * half] * 256u + (unsigned)nin;
  }
#pragma unroll
  for (int nb = 0; nb < 16; ++nb) {
    unsigned n0 = nb * 16;
    int head = nb >> 2;
    v16bf b = load_b_lds(sWB, nb, lane);
    v8f c;
#pragma unroll
    for (int r = 0; r < 8; ++r) c[r] = 0.0f;
    c = wmma32(a, b, c);
    float kv[8], qv[8];
#pragma unroll
    for (int r = 0; r < 8; ++r) kv[r] = k[os[r] + n0];
#pragma unroll
    for (int r = 0; r < 8; ++r) qv[r] = q[od[r] + n0];
#pragma unroll
    for (int r = 0; r < 8; ++r) {
      float p = (c[r] + kv[r]) * qv[r];
      p += __shfl_xor(p, 1); p += __shfl_xor(p, 2);
      p += __shfl_xor(p, 4); p += __shfl_xor(p, 8);
      if (nin == 0) sAcc[wave][(r + 8 * half) * 4 + head] += p;
    }
  }
  for (int i = lane; i < 64; i += 32) {
    int row = i >> 2, h = i & 3;
    float lg = sAcc[wave][i] * 0.125f;  // 1/sqrt(C=64)
    logit[(size_t)(ebase + row) * 4 + h] = lg;
    atomicMax(&nmax[(unsigned)sDst[wave][row] * 4u + (unsigned)h], fkey(lg));
  }
}

__global__ void trans_soft_kernel(float* __restrict__ logit, const unsigned* __restrict__ nmax,
                                  float* __restrict__ nsum, const int* __restrict__ dst, int E_) {
  int t = blockIdx.x * blockDim.x + threadIdx.x;
  if (t >= E_ * 4) return;
  int e = t >> 2, h = t & 3;
  unsigned d = (unsigned)dst[e];
  float m = funkey(nmax[d * 4u + h]);
  if (!__builtin_isfinite(m)) m = 0.0f;
  float ex = __expf(logit[t] - m);
  logit[t] = ex;
  atomicAdd(&nsum[d * 4u + h], ex);
}

__global__ __launch_bounds__(256) void trans_msg_kernel(
    const float* __restrict__ v, const float* __restrict__ ea, const float* __restrict__ we3,
    const int* __restrict__ src, const int* __restrict__ dst,
    const float* __restrict__ ex /*E*4*/, const float* __restrict__ nsum,
    float* __restrict__ out3, int E_) {
  __shared__ unsigned sWB[16 * 32 * 8];
  __shared__ float sEA[8][16 * 32];
  __shared__ int   sSrc[8][16], sDst[8][16];
  __shared__ float sAl[8][64];
  int tid = threadIdx.x;
  stage_b32_frags(we3, 256, sWB, tid, 256);
  __syncthreads();

  int wave = tid >> 5, lane = tid & 31;
  int ebase = (blockIdx.x * 8 + wave) * 16;
  if (ebase >= E_) return;
  for (int i = lane; i < 16 * 32; i += 32) sEA[wave][i] = ea[(size_t)ebase * 32 + i];
  if (lane < 16) { sSrc[wave][lane] = src[ebase + lane]; sDst[wave][lane] = dst[ebase + lane]; }
  for (int i = lane; i < 64; i += 32) {
    int row = i >> 2, h = i & 3;
    unsigned d = (unsigned)sDst[wave][row];
    sAl[wave][i] = ex[(size_t)(ebase + row) * 4 + h] / (nsum[d * 4u + h] + 1e-16f);
  }
  v16bf a = load_a_bf16(sEA[wave], 32, 0, lane);
  int half = lane >> 4, nin = lane & 15;
  unsigned os[8], od[8];
#pragma unroll
  for (int r = 0; r < 8; ++r) {
    os[r] = (unsigned)sSrc[wave][r + 8 * half] * 256u + (unsigned)nin;
    od[r] = (unsigned)sDst[wave][r + 8 * half] * 256u + (unsigned)nin;
  }
  float alh[8];
#pragma unroll
  for (int nb = 0; nb < 16; ++nb) {
    unsigned n0 = nb * 16;
    int head = nb >> 2;
    if ((nb & 3) == 0) {
#pragma unroll
      for (int r = 0; r < 8; ++r) alh[r] = sAl[wave][(r + 8 * half) * 4 + head];
    }
    v16bf b = load_b_lds(sWB, nb, lane);
    v8f c;
#pragma unroll
    for (int r = 0; r < 8; ++r) c[r] = 0.0f;
    c = wmma32(a, b, c);
    float vv[8];
#pragma unroll
    for (int r = 0; r < 8; ++r) vv[r] = v[os[r] + n0];
#pragma unroll
    for (int r = 0; r < 8; ++r)
      atomicAdd(&out3[od[r] + n0], (c[r] + vv[r]) * alh[r]);
  }
}

// ---------------------------------------------------------------------------
// GATv2 pass 1: logit[e] = att . leaky_relu(xl[src]+xr[dst]+ea@we4, 0.2)
// ---------------------------------------------------------------------------
__global__ __launch_bounds__(256) void gat_logit_kernel(
    const float* __restrict__ xl, const float* __restrict__ xr,
    const float* __restrict__ ea, const float* __restrict__ we4 /*32x64*/,
    const float* __restrict__ att, const int* __restrict__ src, const int* __restrict__ dst,
    float* __restrict__ glog, unsigned* __restrict__ gmax, int E_) {
  __shared__ unsigned sWB[4 * 32 * 8];
  __shared__ float sAtt[64];
  __shared__ float sEA[8][16 * 32];
  __shared__ int   sSrc[8][16], sDst[8][16];
  __shared__ float sAcc[8][16];
  int tid = threadIdx.x;
  stage_b32_frags(we4, 64, sWB, tid, 256);
  if (tid < 64) sAtt[tid] = att[tid];
  __syncthreads();

  int wave = tid >> 5, lane = tid & 31;
  int ebase = (blockIdx.x * 8 + wave) * 16;
  if (ebase >= E_) return;
  for (int i = lane; i < 16 * 32; i += 32) sEA[wave][i] = ea[(size_t)ebase * 32 + i];
  if (lane < 16) {
    sSrc[wave][lane] = src[ebase + lane];
    sDst[wave][lane] = dst[ebase + lane];
    sAcc[wave][lane] = 0.0f;
  }
  v16bf a = load_a_bf16(sEA[wave], 32, 0, lane);
  int half = lane >> 4, nin = lane & 15;
  unsigned os[8], od[8];
#pragma unroll
  for (int r = 0; r < 8; ++r) {
    os[r] = (unsigned)sSrc[wave][r + 8 * half] * 64u + (unsigned)nin;
    od[r] = (unsigned)sDst[wave][r + 8 * half] * 64u + (unsigned)nin;
  }
#pragma unroll
  for (int nb = 0; nb < 4; ++nb) {
    unsigned n0 = nb * 16;
    v16bf b = load_b_lds(sWB, nb, lane);
    float av = sAtt[n0 + nin];
    v8f c;
#pragma unroll
    for (int r = 0; r < 8; ++r) c[r] = 0.0f;
    c = wmma32(a, b, c);
    float lv[8], rv[8];
#pragma unroll
    for (int r = 0; r < 8; ++r) lv[r] = xl[os[r] + n0];
#pragma unroll
    for (int r = 0; r < 8; ++r) rv[r] = xr[od[r] + n0];
#pragma unroll
    for (int r = 0; r < 8; ++r) {
      float h = c[r] + lv[r] + rv[r];
      h = (h > 0.0f) ? h : 0.2f * h;
      float p = h * av;
      p += __shfl_xor(p, 1); p += __shfl_xor(p, 2);
      p += __shfl_xor(p, 4); p += __shfl_xor(p, 8);
      if (nin == 0) sAcc[wave][r + 8 * half] += p;
    }
  }
  if (lane < 16) {
    float lg = sAcc[wave][lane];
    glog[ebase + lane] = lg;
    atomicMax(&gmax[(unsigned)sDst[wave][lane]], fkey(lg));
  }
}

__global__ void gat_soft_kernel(float* __restrict__ glog, const unsigned* __restrict__ gmax,
                                float* __restrict__ gsum, const int* __restrict__ dst, int E_) {
  int e = blockIdx.x * blockDim.x + threadIdx.x;
  if (e >= E_) return;
  unsigned d = (unsigned)dst[e];
  float m = funkey(gmax[d]);
  if (!__builtin_isfinite(m)) m = 0.0f;
  float ex = __expf(glog[e] - m);
  glog[e] = ex;
  atomicAdd(&gsum[d], ex);
}

__global__ void gat_out_kernel(const float* __restrict__ xl, const float* __restrict__ glog,
                               const float* __restrict__ gsum, const int* __restrict__ src,
                               const int* __restrict__ dst, float* __restrict__ out4, int E_) {
  int t = blockIdx.x * blockDim.x + threadIdx.x;
  if (t >= E_ * 64) return;
  int e = t >> 6, col = t & 63;
  unsigned s = (unsigned)src[e], d = (unsigned)dst[e];
  float a = glog[e] / (gsum[d] + 1e-16f);
  atomicAdd(&out4[d * 64u + (unsigned)col], xl[s * 64u + (unsigned)col] * a);
}

// ---------------------------------------------------------------------------
// host launcher
// ---------------------------------------------------------------------------
extern "C" void kernel_launch(void* const* d_in, const int* in_sizes, int n_in,
                              void* d_out, int out_size, void* d_ws, size_t ws_size,
                              hipStream_t stream) {
  (void)in_sizes; (void)n_in; (void)out_size; (void)ws_size;
  const int N = 32768, E = 524288;

  const float* x     = (const float*)d_in[0];
  const float* ea    = (const float*)d_in[1];
  const float* eps1  = (const float*)d_in[2];
  const float* w1a   = (const float*)d_in[3];
  const float* b1a   = (const float*)d_in[4];
  const float* w1b   = (const float*)d_in[5];
  const float* b1b   = (const float*)d_in[6];
  const float* we1   = (const float*)d_in[7];
  const float* be1   = (const float*)d_in[8];
  const float* eps2  = (const float*)d_in[9];
  const float* w2a   = (const float*)d_in[10];
  const float* b2a   = (const float*)d_in[11];
  const float* w2b   = (const float*)d_in[12];
  const float* b2b   = (const float*)d_in[13];
  const float* we2   = (const float*)d_in[14];
  const float* be2   = (const float*)d_in[15];
  const float* wq    = (const float*)d_in[16];
  const float* bq    = (const float*)d_in[17];
  const float* wk    = (const float*)d_in[18];
  const float* bk    = (const float*)d_in[19];
  const float* wv    = (const float*)d_in[20];
  const float* bv    = (const float*)d_in[21];
  const float* we3   = (const float*)d_in[22];
  const float* wskip = (const float*)d_in[23];
  const float* bskip = (const float*)d_in[24];
  const float* wl    = (const float*)d_in[25];
  const float* bl    = (const float*)d_in[26];
  const float* wr    = (const float*)d_in[27];
  const float* br    = (const float*)d_in[28];
  const float* we4   = (const float*)d_in[29];
  const float* att   = (const float*)d_in[30];
  const float* bias4 = (const float*)d_in[31];
  const int*   ei    = (const int*)d_in[32];
  const int* srcI = ei;
  const int* dstI = ei + E;

  float* ws = (float*)d_ws;
  size_t off = 0;
  auto take = [&](size_t n) { size_t r = off; off += n; return r; };
  // accumulators first (contiguous, zero-initialized each call)
  size_t AGG   = take((size_t)N * 64);
  size_t OUT3  = take((size_t)N * 256);
  size_t OUT4  = take((size_t)N * 64);
  size_t NMAX4 = take((size_t)N * 4);
  size_t NSUM4 = take((size_t)N * 4);
  size_t GMAX  = take((size_t)N);
  size_t GSUM  = take((size_t)N);
  size_t ZTOT  = off;
  size_t HBUF  = take((size_t)N * 64);
  size_t HID   = take((size_t)N * 64);
  size_t X1    = take((size_t)N * 64);
  size_t X2    = take((size_t)N * 64);
  size_t Q     = take((size_t)N * 256);
  size_t Kb    = take((size_t)N * 256);
  size_t Vb    = take((size_t)N * 256);
  size_t SKIP  = take((size_t)N * 256);
  size_t X3    = take((size_t)N * 256);
  size_t XL    = take((size_t)N * 64);
  size_t XR    = take((size_t)N * 64);
  size_t TLOG  = take((size_t)E * 4);
  size_t GLOG  = take((size_t)E);

  dim3 blk(256);
  auto gridOf = [](int n) { return dim3((unsigned)((n + 255) / 256)); };
  dim3 egrid(E / 128);   // 16 edges/wave, 8 waves/block
  dim3 ngrid8(N / 128);  // 16 rows/wave, 8 waves/block

  // 0. zero accumulators
  zero_kernel<<<gridOf((int)ZTOT), blk, 0, stream>>>(ws, (int)ZTOT);

  // --- GINEConv 1 ---
  gine_edge_kernel<<<egrid, blk, 0, stream>>>(x, ea, we1, be1, srcI, dstI, ws + AGG, E);
  gine_h_kernel<<<gridOf(N * 64), blk, 0, stream>>>(x, ws + AGG, eps1, ws + HBUF, N * 64);
  node_linear_kernel<64, 8><<<ngrid8, blk, 0, stream>>>(ws + HBUF, w1a, b1a, ws + HID, 64, 1, N);
  node_linear_kernel<64, 8><<<ngrid8, blk, 0, stream>>>(ws + HID, w1b, b1b, ws + X1, 64, 0, N);

  // --- GINEConv 2 ---
  zero_kernel<<<gridOf(N * 64), blk, 0, stream>>>(ws + AGG, N * 64);
  gine_edge_kernel<<<egrid, blk, 0, stream>>>(ws + X1, ea, we2, be2, srcI, dstI, ws + AGG, E);
  gine_h_kernel<<<gridOf(N * 64), blk, 0, stream>>>(ws + X1, ws + AGG, eps2, ws + HBUF, N * 64);
  node_linear_kernel<64, 8><<<ngrid8, blk, 0, stream>>>(ws + HBUF, w2a, b2a, ws + HID, 32, 1, N);
  node_linear_kernel<32, 8><<<ngrid8, blk, 0, stream>>>(ws + HID, w2b, b2b, ws + X2, 64, 0, N);

  // --- TransformerConv ---
  node_linear_kernel<64, 8><<<ngrid8, blk, 0, stream>>>(ws + X2, wq, bq, ws + Q, 256, 0, N);
  node_linear_kernel<64, 8><<<ngrid8, blk, 0, stream>>>(ws + X2, wk, bk, ws + Kb, 256, 0, N);
  node_linear_kernel<64, 8><<<ngrid8, blk, 0, stream>>>(ws + X2, wv, bv, ws + Vb, 256, 0, N);
  node_linear_kernel<64, 8><<<ngrid8, blk, 0, stream>>>(ws + X2, wskip, bskip, ws + SKIP, 256, 0, N);
  trans_logit_kernel<<<egrid, blk, 0, stream>>>(ws + Q, ws + Kb, ea, we3, srcI, dstI,
                                                ws + TLOG, (unsigned*)(ws + NMAX4), E);
  trans_soft_kernel<<<gridOf(E * 4), blk, 0, stream>>>(ws + TLOG, (const unsigned*)(ws + NMAX4),
                                                       ws + NSUM4, dstI, E);
  trans_msg_kernel<<<egrid, blk, 0, stream>>>(ws + Vb, ea, we3, srcI, dstI,
                                              ws + TLOG, ws + NSUM4, ws + OUT3, E);
  add_kernel<<<gridOf(N * 256), blk, 0, stream>>>(ws + OUT3, ws + SKIP, ws + X3, N * 256);

  // --- GATv2Conv ---
  node_linear_kernel<256, 2><<<dim3(N / 32), dim3(64), 0, stream>>>(ws + X3, wl, bl, ws + XL, 64, 0, N);
  node_linear_kernel<256, 2><<<dim3(N / 32), dim3(64), 0, stream>>>(ws + X3, wr, br, ws + XR, 64, 0, N);
  gat_logit_kernel<<<egrid, blk, 0, stream>>>(ws + XL, ws + XR, ea, we4, att, srcI, dstI,
                                              ws + GLOG, (unsigned*)(ws + GMAX), E);
  gat_soft_kernel<<<gridOf(E), blk, 0, stream>>>(ws + GLOG, (const unsigned*)(ws + GMAX),
                                                 ws + GSUM, dstI, E);
  gat_out_kernel<<<dim3(E * 64 / 256), blk, 0, stream>>>(ws + XL, ws + GLOG, ws + GSUM,
                                                         srcI, dstI, ws + OUT4, E);
  bias_out_kernel<<<gridOf(N * 64), blk, 0, stream>>>(ws + OUT4, bias4, (float*)d_out, N * 64);
}
